// PN2Encoder_60043642798275
// MI455X (gfx1250) — compile-verified, matrix operand order checked by reading, not run on hardware
//
#include <hip/hip_runtime.h>
#include <hip/hip_bf16.h>

typedef __attribute__((ext_vector_type(16))) _Float16 v16h;
typedef __attribute__((ext_vector_type(8)))  float    v8f;

// K-index mapping for 16-bit A/B WMMA fragments (CDNA5 ISA 7.12.2):
// element e (0..15) of the v16h, lane-half g (0 = lanes 0-15, 1 = lanes 16-31).
__device__ __forceinline__ int kmap(int e, int g) {
  int v = e >> 1;
  return ((v & 4) << 2) + ((v & 3) << 1) + (e & 1) + (g << 3);
}

// ---------------------------------------------------------------------------
// Weight pre-swizzle: f32 [KACT x COUT] -> f16 fragments laid out so that
// B-frag(kt,nt) for lane L is 16 contiguous halfs (one 32B v16h load).
// ---------------------------------------------------------------------------
__global__ void __launch_bounds__(256) wswz_kernel(const float* __restrict__ W,
                                                   _Float16* __restrict__ out,
                                                   int KACT, int KPAD, int COUT) {
  int t = blockIdx.x * blockDim.x + threadIdx.x;
  int total = KPAD * COUT;
  if (t >= total) return;
  int frag = t >> 9;         // 512 halfs per 32x16 fragment
  int win  = t & 511;
  int lane = win >> 4;
  int e    = win & 15;
  int ntiles = COUT >> 4;
  int nt = frag % ntiles;
  int kt = frag / ntiles;
  int g  = lane >> 4, li = lane & 15;
  int k  = kt * 32 + kmap(e, g);
  int n  = nt * 16 + li;
  float v = (k < KACT) ? W[(size_t)k * COUT + n] : 0.0f;
  out[t] = (_Float16)v;
}

// ---------------------------------------------------------------------------
// Farthest point sampling: one block per batch, serial argmax iterations.
// Writes selected positions directly (indices not needed downstream).
// ---------------------------------------------------------------------------
template<int NMAX>
__global__ void __launch_bounds__(256) fps_kernel(const float* __restrict__ pos,
                                                  int N, int M,
                                                  float* __restrict__ selpos) {
  __shared__ float d[NMAX];
  __shared__ float rv[256];
  __shared__ int   ri[256];
  const int b = blockIdx.x, tid = threadIdx.x;
  const float* P = pos + (size_t)b * N * 3;
  float px = P[0], py = P[1], pz = P[2];
  if (tid == 0) {
    selpos[(size_t)b * M * 3 + 0] = px;
    selpos[(size_t)b * M * 3 + 1] = py;
    selpos[(size_t)b * M * 3 + 2] = pz;
  }
  for (int j = tid; j < N; j += 256) {
    float dx = P[j * 3 + 0] - px, dy = P[j * 3 + 1] - py, dz = P[j * 3 + 2] - pz;
    d[j] = dx * dx + dy * dy + dz * dz;
  }
  __syncthreads();
  for (int it = 1; it < M; ++it) {
    float bv = -1.0f; int bi = N;
    for (int j = tid; j < N; j += 256) {
      float v = d[j];
      if (v > bv) { bv = v; bi = j; }   // ascending j -> first max per thread
    }
    rv[tid] = bv; ri[tid] = bi;
    __syncthreads();
    for (int s = 128; s > 0; s >>= 1) {
      if (tid < s) {
        float ov = rv[tid + s]; int oi = ri[tid + s];
        if (ov > rv[tid] || (ov == rv[tid] && oi < ri[tid])) { rv[tid] = ov; ri[tid] = oi; }
      }
      __syncthreads();
    }
    int i = ri[0];
    float cx = P[i * 3 + 0], cy = P[i * 3 + 1], cz = P[i * 3 + 2];
    if (tid == 0) {
      selpos[((size_t)b * M + it) * 3 + 0] = cx;
      selpos[((size_t)b * M + it) * 3 + 1] = cy;
      selpos[((size_t)b * M + it) * 3 + 2] = cz;
    }
    for (int j = tid; j < N; j += 256) {
      float dx = P[j * 3 + 0] - cx, dy = P[j * 3 + 1] - cy, dz = P[j * 3 + 2] - cz;
      d[j] = fminf(d[j], dx * dx + dy * dy + dz * dz);
    }
    __syncthreads();
  }
}

// ---------------------------------------------------------------------------
// Ball query: one block per (batch,center); 64 serial argmin selections of
// the nearest points within radius (r2). Matches top_k(-d2) tie-breaking.
// ---------------------------------------------------------------------------
template<int NMAX>
__global__ void __launch_bounds__(256) ball_kernel(const float* __restrict__ pos,
                                                   const float* __restrict__ selpos,
                                                   int N, int M, float r2,
                                                   int* __restrict__ nidx,
                                                   int* __restrict__ valid) {
  __shared__ float d2[NMAX];
  __shared__ float rv[256];
  __shared__ int   ri[256];
  const int bm = blockIdx.x, b = bm / M, tid = threadIdx.x;
  const float INF = __builtin_inff();
  const float* P = pos + (size_t)b * N * 3;
  float cx = selpos[(size_t)bm * 3 + 0];
  float cy = selpos[(size_t)bm * 3 + 1];
  float cz = selpos[(size_t)bm * 3 + 2];
  for (int j = tid; j < N; j += 256) {
    float dx = P[j * 3 + 0] - cx, dy = P[j * 3 + 1] - cy, dz = P[j * 3 + 2] - cz;
    float dd = dx * dx + dy * dy + dz * dz;
    d2[j] = (dd <= r2) ? dd : INF;
  }
  __syncthreads();
  for (int kk = 0; kk < 64; ++kk) {
    float bv = INF; int bi = N;
    for (int j = tid; j < N; j += 256) {
      float v = d2[j];
      if (v < bv || (v == bv && j < bi)) { bv = v; bi = j; }
    }
    rv[tid] = bv; ri[tid] = bi;
    __syncthreads();
    for (int s = 128; s > 0; s >>= 1) {
      if (tid < s) {
        float ov = rv[tid + s]; int oi = ri[tid + s];
        if (ov < rv[tid] || (ov == rv[tid] && oi < ri[tid])) { rv[tid] = ov; ri[tid] = oi; }
      }
      __syncthreads();
    }
    if (tid == 0) {
      int i = ri[0]; float mv = rv[0];
      nidx [(size_t)bm * 64 + kk] = i;
      valid[(size_t)bm * 64 + kk] = (mv < INF) ? 1 : 0;
      d2[i] = INF;   // remove from candidate set
    }
    __syncthreads();
  }
}

// ---------------------------------------------------------------------------
// One MLP layer on a per-wave 16-row tile using v_wmma_f32_16x16x32_f16.
// Ain: f16 activations in LDS (row stride LDIN). Wz: pre-swizzled f16 weights.
// Non-final: relu -> f16 -> Aout (row stride LDOUT).
// Final: relu -> masked column-max -> int-bits atomicMax into colmax.
// ---------------------------------------------------------------------------
template<int KPAD, int COUT, int LDIN, int LDOUT, bool FINAL>
__device__ __forceinline__ void mlp_layer(const _Float16* __restrict__ Ain,
                                          const _Float16* __restrict__ Wz,
                                          const float* __restrict__ bias,
                                          _Float16* __restrict__ Aout,
                                          int* __restrict__ colmax,
                                          const int* __restrict__ vflag,
                                          int wave, int lane) {
  const int g = lane >> 4, li = lane & 15;
  const int row0 = wave * 16;
  const v16h* Wf = (const v16h*)Wz;
  for (int nt = 0; nt < COUT / 16; ++nt) {
    const int n = nt * 16 + li;
    v8f acc = {};
    for (int kt = 0; kt < KPAD / 32; ++kt) {
      v16h a;
#pragma unroll
      for (int e = 0; e < 16; ++e)
        a[e] = Ain[(row0 + li) * LDIN + kt * 32 + kmap(e, g)];
      v16h bf = Wf[(kt * (COUT / 16) + nt) * 32 + lane];
      acc = __builtin_amdgcn_wmma_f32_16x16x32_f16(false, a, false, bf,
                                                   (short)0, acc, false, false);
    }
    const float bn = bias[n];
    if constexpr (FINAL) {
      float mx = 0.0f;   // post-ReLU values >= 0 and >=1 valid row exists
#pragma unroll
      for (int j = 0; j < 8; ++j) {
        const int r = row0 + j + 8 * g;     // C/D layout: VGPR j -> row j+8g
        float v = fmaxf(acc[j] + bn, 0.0f);
        if (vflag[r]) mx = fmaxf(mx, v);
      }
      mx = fmaxf(mx, __shfl_xor(mx, 16, 32));
      if (g == 0) atomicMax(&colmax[n], __float_as_int(mx));
    } else {
#pragma unroll
      for (int j = 0; j < 8; ++j) {
        const int r = row0 + j + 8 * g;
        float v = fmaxf(acc[j] + bn, 0.0f);
        Aout[r * LDOUT + n] = (_Float16)v;
      }
    }
  }
}

// ---------------------------------------------------------------------------
// Fused set-abstraction block: gather K=64 neighbors -> 3-layer MLP (WMMA)
// -> masked max over neighbors. One block (4 waves) per (batch,center).
// ---------------------------------------------------------------------------
template<int CFEAT, int CINP, int C1, int C2, int C3, int WMAX>
__global__ void __launch_bounds__(128) sa_mlp_kernel(
    const float* __restrict__ xin, const float* __restrict__ pos,
    const float* __restrict__ selpos, const int* __restrict__ nidx,
    const int* __restrict__ valid,
    const _Float16* __restrict__ Wz0, const float* __restrict__ b0,
    const _Float16* __restrict__ Wz1, const float* __restrict__ b1,
    const _Float16* __restrict__ Wz2, const float* __restrict__ b2,
    float* __restrict__ xout, int Npts, int M) {
  __shared__ _Float16 bufA[64 * WMAX];
  __shared__ _Float16 bufB[64 * WMAX];
  __shared__ int colmax[C3];
  __shared__ int vflag[64];
  const int tid = threadIdx.x;
  const int bm = blockIdx.x, b = bm / M;
  const int* ni = &nidx[(size_t)bm * 64];
  const float sx = selpos[(size_t)bm * 3 + 0];
  const float sy = selpos[(size_t)bm * 3 + 1];
  const float sz = selpos[(size_t)bm * 3 + 2];
  for (int r = tid; r < 64; r += 128) vflag[r] = valid[(size_t)bm * 64 + r];
  for (int c = tid; c < C3; c += 128) colmax[c] = 0;
  // gather [xj | pj - center | zero-pad] -> f16 LDS
  for (int t = tid; t < 64 * CINP; t += 128) {
    int row = t / CINP, col = t % CINP;
    int src = ni[row];
    float v = 0.0f;
    if (col < CFEAT) {
      v = xin[((size_t)b * Npts + src) * CFEAT + col];
    } else if (col < CFEAT + 3) {
      int c = col - CFEAT;
      float pv = pos[((size_t)b * Npts + src) * 3 + c];
      v = pv - (c == 0 ? sx : (c == 1 ? sy : sz));
    }
    bufA[row * WMAX + col] = (_Float16)v;
  }
  __syncthreads();
  const int wave = tid >> 5, lane = tid & 31;
  mlp_layer<CINP, C1, WMAX, WMAX, false>(bufA, Wz0, b0, bufB, nullptr, nullptr, wave, lane);
  mlp_layer<C1,   C2, WMAX, WMAX, false>(bufB, Wz1, b1, bufA, nullptr, nullptr, wave, lane);
  mlp_layer<C2,   C3, WMAX, 1,    true >(bufA, Wz2, b2, nullptr, colmax, vflag, wave, lane);
  __syncthreads();
  for (int c = tid; c < C3; c += 128)
    xout[(size_t)bm * C3 + c] = __int_as_float(colmax[c]);
}

// ---------------------------------------------------------------------------
// Global MLP + global max pool. 32 rows (2 waves) per block, 7 blocks/batch.
// Folds into d_out with global int-bits atomicMax (values >= 0 post-ReLU).
// ---------------------------------------------------------------------------
__global__ void __launch_bounds__(64) gsa_kernel(
    const float* __restrict__ x2, const float* __restrict__ pos2,
    const _Float16* __restrict__ Wz0, const float* __restrict__ b0,
    const _Float16* __restrict__ Wz1, const float* __restrict__ b1,
    const _Float16* __restrict__ Wz2, const float* __restrict__ b2,
    float* __restrict__ out) {
  constexpr int ROWS = 32, CINP = 288, C3 = 1024;
  __shared__ _Float16 bufA[ROWS * 512];   // holds 288-wide input, then 512-wide act
  __shared__ _Float16 bufB[ROWS * 256];   // holds 256-wide act
  __shared__ int colmax[C3];
  __shared__ int vflag[ROWS];
  const int tid = threadIdx.x;
  const int b = blockIdx.x / 7, blk = blockIdx.x % 7;
  const int row0g = blk * ROWS;
  for (int r = tid; r < ROWS; r += 64) vflag[r] = ((row0g + r) < 200) ? 1 : 0;
  for (int c = tid; c < C3; c += 64) colmax[c] = 0;
  for (int t = tid; t < ROWS * CINP; t += 64) {
    int row = t / CINP, col = t % CINP;
    int rg = row0g + row;
    float v = 0.0f;
    if (rg < 200) {
      if (col < 256)      v = x2 [((size_t)b * 200 + rg) * 256 + col];
      else if (col < 259) v = pos2[((size_t)b * 200 + rg) * 3 + (col - 256)];
    }
    bufA[row * 512 + col] = (_Float16)v;
  }
  __syncthreads();
  const int wave = tid >> 5, lane = tid & 31;
  mlp_layer<288, 256,  512, 256, false>(bufA, Wz0, b0, bufB, nullptr, nullptr, wave, lane);
  mlp_layer<256, 512,  256, 512, false>(bufB, Wz1, b1, bufA, nullptr, nullptr, wave, lane);
  mlp_layer<512, 1024, 512, 1,   true >(bufA, Wz2, b2, nullptr, colmax, vflag, wave, lane);
  __syncthreads();
  for (int c = tid; c < C3; c += 64)
    atomicMax((int*)out + (size_t)b * 1024 + c, colmax[c]);
}

__global__ void __launch_bounds__(256) zero_kernel(float* __restrict__ p, int n) {
  int i = blockIdx.x * blockDim.x + threadIdx.x;
  if (i < n) p[i] = 0.0f;
}

// ---------------------------------------------------------------------------
extern "C" void kernel_launch(void* const* d_in, const int* in_sizes, int n_in,
                              void* d_out, int out_size, void* d_ws, size_t ws_size,
                              hipStream_t stream) {
  (void)in_sizes; (void)n_in; (void)out_size; (void)ws_size;
  const float* x     = (const float*)d_in[0];
  const float* pos   = (const float*)d_in[1];
  const float* Wsrc[9] = {
    (const float*)d_in[2],  (const float*)d_in[4],  (const float*)d_in[6],   // sa1
    (const float*)d_in[8],  (const float*)d_in[10], (const float*)d_in[12],  // sa2
    (const float*)d_in[14], (const float*)d_in[16], (const float*)d_in[18],  // gsa
  };
  const float* sa1b0 = (const float*)d_in[3];
  const float* sa1b1 = (const float*)d_in[5];
  const float* sa1b2 = (const float*)d_in[7];
  const float* sa2b0 = (const float*)d_in[9];
  const float* sa2b1 = (const float*)d_in[11];
  const float* sa2b2 = (const float*)d_in[13];
  const float* gsab0 = (const float*)d_in[15];
  const float* gsab1 = (const float*)d_in[17];
  const float* gsab2 = (const float*)d_in[19];

  char* ws = (char*)d_ws;
  size_t off = 0;
  auto alloc = [&](size_t bytes) -> void* {
    void* p = ws + off;
    off += (bytes + 255) & ~(size_t)255;
    return p;
  };

  // weight swizzle (KACT -> zero-padded KPAD, WMMA B-fragment order, f16)
  const int KACT[9] = { 6,  64, 64, 131, 128, 128, 259, 256, 512 };
  const int KPAD[9] = { 32, 64, 64, 160, 128, 128, 288, 256, 512 };
  const int COUT[9] = { 64, 64, 128, 128, 128, 256, 256, 512, 1024 };
  _Float16* wz[9];
  for (int i = 0; i < 9; ++i) {
    int total = KPAD[i] * COUT[i];
    wz[i] = (_Float16*)alloc((size_t)total * sizeof(_Float16));
    wswz_kernel<<<(total + 255) / 256, 256, 0, stream>>>(Wsrc[i], wz[i],
                                                         KACT[i], KPAD[i], COUT[i]);
  }

  float* pos1  = (float*)alloc((size_t)16 * 800 * 3   * 4);
  int*   nidx1 = (int*)  alloc((size_t)16 * 800 * 64  * 4);
  int*   val1  = (int*)  alloc((size_t)16 * 800 * 64  * 4);
  float* x1    = (float*)alloc((size_t)16 * 800 * 128 * 4);
  float* pos2  = (float*)alloc((size_t)16 * 200 * 3   * 4);
  int*   nidx2 = (int*)  alloc((size_t)16 * 200 * 64  * 4);
  int*   val2  = (int*)  alloc((size_t)16 * 200 * 64  * 4);
  float* x2    = (float*)alloc((size_t)16 * 200 * 256 * 4);

  // SA1: N=4000 -> M=800, r=0.2, K=64, dims [6,64,64,128]
  fps_kernel<4000><<<16, 256, 0, stream>>>(pos, 4000, 800, pos1);
  ball_kernel<4000><<<16 * 800, 256, 0, stream>>>(pos, pos1, 4000, 800, 0.04f,
                                                  nidx1, val1);
  sa_mlp_kernel<3, 32, 64, 64, 128, 64><<<16 * 800, 128, 0, stream>>>(
      x, pos, pos1, nidx1, val1,
      wz[0], sa1b0, wz[1], sa1b1, wz[2], sa1b2, x1, 4000, 800);

  // SA2: N=800 -> M=200, r=0.4, K=64, dims [131,128,128,256]
  fps_kernel<800><<<16, 256, 0, stream>>>(pos1, 800, 200, pos2);
  ball_kernel<800><<<16 * 200, 256, 0, stream>>>(pos1, pos2, 800, 200, 0.16f,
                                                 nidx2, val2);
  sa_mlp_kernel<128, 160, 128, 128, 256, 160><<<16 * 200, 128, 0, stream>>>(
      x1, pos1, pos2, nidx2, val2,
      wz[3], sa2b0, wz[4], sa2b1, wz[5], sa2b2, x2, 800, 200);

  // GSA: [259,256,512,1024] + global max pool -> d_out [16,1024]
  zero_kernel<<<64, 256, 0, stream>>>((float*)d_out, 16 * 1024);
  gsa_kernel<<<16 * 7, 64, 0, stream>>>(x2, pos2,
                                        wz[6], gsab0, wz[7], gsab1, wz[8], gsab2,
                                        (float*)d_out);
}
